// MambaBlock_4303557230699
// MI455X (gfx1250) — matrix operand order repaired and checked by reading, not compiled, not run on hardware
//
#include <hip/hip_runtime.h>
#include <hip/hip_bf16.h>

// ---------------------------------------------------------------------------
// Mamba block for MI455X (gfx1250, wave32).
// GEMMs run on v_wmma_f32_16x16x32_bf16 (f32 accumulate) with
// global_load_async_to_lds_b128 staging (ASYNCcnt-tracked, no VGPR staging).
// Elementwise + selective-scan in f32.
// ---------------------------------------------------------------------------

typedef __attribute__((ext_vector_type(16))) __bf16 v16bf;
typedef __attribute__((ext_vector_type(8)))  __bf16 v8bf;
typedef __attribute__((ext_vector_type(8)))  float  v8f;

#define BM 128
#define BN 128
#define BK 32
#define LDK 40   // padded LDS row (bf16 elems): 80B rows -> conflict-free b128

// CDNA5 async global->LDS copy, 16B per lane, tracked with ASYNCcnt.
__device__ __forceinline__ void async_ld_b128(unsigned lds_addr, const void* gaddr) {
  asm volatile("global_load_async_to_lds_b128 %0, %1, off"
               :: "v"(lds_addr), "v"(gaddr)
               : "memory");
}
__device__ __forceinline__ void wait_async0() {
  asm volatile("s_wait_asynccnt 0" ::: "memory");
}

// ---------------------------------------------------------------------------
// f32 -> bf16 conversion
// ---------------------------------------------------------------------------
__global__ void cvt_f32_bf16_kernel(const float* __restrict__ in,
                                    __bf16* __restrict__ out, int n) {
  int i = blockIdx.x * blockDim.x + threadIdx.x;
  if (i < n) out[i] = (__bf16)in[i];
}

// ---------------------------------------------------------------------------
// LayerNorm over last dim (1024), one block per token, bf16 output
// ---------------------------------------------------------------------------
__global__ void ln_bf16_kernel(const float* __restrict__ x,
                               const float* __restrict__ w,
                               const float* __restrict__ b,
                               __bf16* __restrict__ out) {
  __shared__ float s_sum[256];
  __shared__ float s_sq[256];
  const int row = blockIdx.x;
  const int t = threadIdx.x;
  const float* xr = x + (size_t)row * 1024;
  float v[4];
  float sum = 0.f, sq = 0.f;
#pragma unroll
  for (int i = 0; i < 4; ++i) {
    v[i] = xr[t + 256 * i];
    sum += v[i];
    sq += v[i] * v[i];
  }
  s_sum[t] = sum; s_sq[t] = sq;
  __syncthreads();
  for (int off = 128; off > 0; off >>= 1) {
    if (t < off) { s_sum[t] += s_sum[t + off]; s_sq[t] += s_sq[t + off]; }
    __syncthreads();
  }
  const float mu  = s_sum[0] * (1.f / 1024.f);
  const float var = s_sq[0] * (1.f / 1024.f) - mu * mu;
  const float inv = rsqrtf(var + 1e-5f);
#pragma unroll
  for (int i = 0; i < 4; ++i) {
    int c = t + 256 * i;
    out[(size_t)row * 1024 + c] = (__bf16)((v[i] - mu) * inv * w[c] + b[c]);
  }
}

// ---------------------------------------------------------------------------
// bf16 GEMM: C[M,N] = A[M,K] * B[N,K]^T (+bias[N]) (+residual[M,N]), f32 acc.
// 256 threads (8 waves, 2x4), 128x128x32 tiles, double-buffered LDS filled by
// global_load_async_to_lds_b128, compute on v_wmma_f32_16x16x32_bf16.
// M must be a multiple of 128, K of 32.
// ---------------------------------------------------------------------------
__global__ __launch_bounds__(256, 2)
void gemm_bf16_kernel(const __bf16* __restrict__ A, const __bf16* __restrict__ B,
                      float* __restrict__ C, int M, int N, int K,
                      const float* __restrict__ bias,
                      const float* __restrict__ residual) {
  __shared__ __bf16 As[2][BM][LDK];
  __shared__ __bf16 Bs[2][BN][LDK];

  const int t    = threadIdx.x;
  const int lane = t & 31;
  const int wave = t >> 5;
  const int wm   = wave >> 2;  // 0..1 : 64 rows each
  const int wn   = wave & 3;   // 0..3 : 32 cols each
  const int blockN0 = blockIdx.x * BN;
  const int blockM0 = blockIdx.y * BM;

  // global->LDS staging: each thread moves 16 bf16 of A and 16 of B per k-step
  const int lrow = t >> 1;        // 0..127
  const int lk   = (t & 1) * 16;  // 0 or 16
  const __bf16* gA = A + (size_t)(blockM0 + lrow) * K + lk;
  const int gBrow  = blockN0 + lrow;
  // OOB B rows alias row 0; their WMMA output columns are masked at the store.
  const __bf16* gB = B + (size_t)(gBrow < N ? gBrow : 0) * K + lk;

  v8f zacc = {};
  v8f acc[4][2];
#pragma unroll
  for (int mt = 0; mt < 4; ++mt)
#pragma unroll
    for (int nt = 0; nt < 2; ++nt) acc[mt][nt] = zacc;

  const int nk = K / BK;

  // prologue: issue async fills for stage 0
  {
    const unsigned la = (unsigned)(size_t)&As[0][lrow][lk];
    const unsigned lb = (unsigned)(size_t)&Bs[0][lrow][lk];
    async_ld_b128(la,      gA);
    async_ld_b128(la + 16, gA + 8);
    async_ld_b128(lb,      gB);
    async_ld_b128(lb + 16, gB + 8);
  }

  const int frow  = lane & 15;   // row/col within 16x16 tile
  const int fhalf = lane >> 4;   // half-wave selects K-subgroup

  for (int kt = 0; kt < nk; ++kt) {
    const int s = kt & 1;

    wait_async0();       // this wave's fills of stage s have landed in LDS
    __syncthreads();     // all waves' fills landed; prior reads of s^1 done

    if (kt + 1 < nk) {   // issue fills of the other stage; overlap with compute
      const __bf16* pA = gA + (size_t)(kt + 1) * BK;
      const __bf16* pB = gB + (size_t)(kt + 1) * BK;
      const unsigned la = (unsigned)(size_t)&As[s ^ 1][lrow][lk];
      const unsigned lb = (unsigned)(size_t)&Bs[s ^ 1][lrow][lk];
      async_ld_b128(la,      pA);
      async_ld_b128(la + 16, pA + 8);
      async_ld_b128(lb,      pB);
      async_ld_b128(lb + 16, pB + 8);
      if (kt + 2 < nk) {
        __builtin_prefetch(gA + (size_t)(kt + 2) * BK, 0, 1);  // global_prefetch_b8
        __builtin_prefetch(gB + (size_t)(kt + 2) * BK, 0, 1);
      }
    }

    // Build fragments per ISA 7.12.2 VGPR layouts (wave32).
    v16bf afr[4], bfr[2];
#pragma unroll
    for (int mt = 0; mt < 4; ++mt) {
      const int r = wm * 64 + mt * 16 + frow;
      v8bf lo = *(const v8bf*)&As[s][r][fhalf * 8];        // K 0-7  / 8-15
      v8bf hi = *(const v8bf*)&As[s][r][16 + fhalf * 8];   // K 16-23/24-31
#pragma unroll
      for (int i = 0; i < 8; ++i) { afr[mt][i] = lo[i]; afr[mt][8 + i] = hi[i]; }
    }
#pragma unroll
    for (int nt = 0; nt < 2; ++nt) {
      const int c = wn * 32 + nt * 16 + frow;
      v8bf lo = *(const v8bf*)&Bs[s][c][fhalf * 8];        // K 0-7  / 8-15
      v8bf hi = *(const v8bf*)&Bs[s][c][16 + fhalf * 8];   // K 16-23/24-31
#pragma unroll
      for (int i = 0; i < 8; ++i) { bfr[nt][i] = lo[i]; bfr[nt][8 + i] = hi[i]; }
    }

#pragma unroll
    for (int mt = 0; mt < 4; ++mt)
#pragma unroll
      for (int nt = 0; nt < 2; ++nt)
        acc[mt][nt] = __builtin_amdgcn_wmma_f32_16x16x32_bf16(
            false, afr[mt], false, bfr[nt], (short)0, acc[mt][nt], false, false);
  }

  // epilogue: C/D layout — VGPR r: lanes0-15 -> M=r, lanes16-31 -> M=r+8
  const int rhalf = (lane >> 4) * 8;
#pragma unroll
  for (int mt = 0; mt < 4; ++mt) {
#pragma unroll
    for (int nt = 0; nt < 2; ++nt) {
      const int col = blockN0 + wn * 32 + nt * 16 + frow;
      if (col >= N) continue;
      const float bv = bias ? bias[col] : 0.0f;
#pragma unroll
      for (int r = 0; r < 8; ++r) {
        const int row = blockM0 + wm * 64 + mt * 16 + r + rhalf;
        const size_t idx = (size_t)row * N + col;
        float v = acc[mt][nt][r] + bv;
        if (residual) v += residual[idx];
        C[idx] = v;
      }
    }
  }
}

// ---------------------------------------------------------------------------
// Causal depthwise conv1d (taps=4) + bias + SiLU on xz[..., :2048]
// ---------------------------------------------------------------------------
__global__ void conv_silu_kernel(const float* __restrict__ xz,
                                 const float* __restrict__ cw,
                                 const float* __restrict__ cb,
                                 float* __restrict__ xc_f,
                                 __bf16* __restrict__ xc_bf,
                                 int L, int Etot) {
  const int idx = blockIdx.x * blockDim.x + threadIdx.x;
  const int total = 2 * L * Etot;
  if (idx >= total) return;
  const int e = idx % Etot;
  const int l = (idx / Etot) % L;
  const int b = idx / (Etot * L);
  float acc = cb[e];
#pragma unroll
  for (int j = 0; j < 4; ++j) {
    const int ll = l - 3 + j;
    if (ll >= 0)
      acc += cw[e * 4 + j] * xz[((size_t)b * L + ll) * 4096 + e];
  }
  const float s = acc / (1.f + __expf(-acc));  // SiLU
  const size_t o = ((size_t)b * L + l) * 2048 + e;
  xc_f[o] = s;
  xc_bf[o] = (__bf16)s;
}

// ---------------------------------------------------------------------------
// extract dt columns (first 64 of xdbc) as bf16
// ---------------------------------------------------------------------------
__global__ void extract_dt_kernel(const float* __restrict__ xdbc,
                                  __bf16* __restrict__ dt_bf, int Mrows) {
  const int i = blockIdx.x * blockDim.x + threadIdx.x;
  if (i >= Mrows * 64) return;
  const int r = i >> 6, c = i & 63;
  dt_bf[i] = (__bf16)xdbc[(size_t)r * 96 + c];
}

__global__ void softplus_kernel(float* __restrict__ d, int n) {
  const int i = blockIdx.x * blockDim.x + threadIdx.x;
  if (i >= n) return;
  const float x = d[i];
  d[i] = (x > 20.f) ? x : log1pf(__expf(x));
}

// ---------------------------------------------------------------------------
// Selective scan + skip + SiLU(z) gating. Block = 16 channels x 16 states.
// Recurrence per (b,e,n) lane; 16-lane shfl_xor reduction over n (wave32).
// ---------------------------------------------------------------------------
__global__ __launch_bounds__(256)
void scan_kernel(const float* __restrict__ delta,
                 const float* __restrict__ xc,
                 const float* __restrict__ xdbc,   // [M,96]: 64..79=B, 80..95=C
                 const float* __restrict__ xz,     // z at cols 2048..4095
                 const float* __restrict__ A_log,
                 const float* __restrict__ Dp,
                 __bf16* __restrict__ y_bf, int L) {
  const int b    = blockIdx.x >> 7;        // 2048/16 = 128 blocks per batch
  const int eblk = blockIdx.x & 127;
  const int t = threadIdx.x;
  const int n = t & 15;
  const int e = eblk * 16 + (t >> 4);
  const float Av = -__expf(A_log[(size_t)e * 16 + n]);
  const float Dv = Dp[e];
  float h = 0.f;
  for (int l = 0; l < L; ++l) {
    const size_t row = (size_t)b * L + l;
    const float dlt = delta[row * 2048 + e];
    const float xcv = xc[row * 2048 + e];
    const float Bv = xdbc[row * 96 + 64 + n];
    const float Cv = xdbc[row * 96 + 80 + n];
    h = __expf(dlt * Av) * h + dlt * Bv * xcv;
    float p = h * Cv;
    p += __shfl_xor(p, 1, 32);
    p += __shfl_xor(p, 2, 32);
    p += __shfl_xor(p, 4, 32);
    p += __shfl_xor(p, 8, 32);
    if (n == 0) {
      float y = p + xcv * Dv;
      const float z = xz[row * 4096 + 2048 + e];
      y *= z / (1.f + __expf(-z));
      y_bf[row * 2048 + e] = (__bf16)y;
    }
  }
}

__global__ void gelu_bf16_kernel(const float* __restrict__ g,
                                 __bf16* __restrict__ out, int n) {
  const int i = blockIdx.x * blockDim.x + threadIdx.x;
  if (i >= n) return;
  const float x = g[i];
  out[i] = (__bf16)(0.5f * x * (1.f + erff(x * 0.70710678f)));
}

// ---------------------------------------------------------------------------
extern "C" void kernel_launch(void* const* d_in, const int* in_sizes, int n_in,
                              void* d_out, int out_size, void* d_ws, size_t ws_size,
                              hipStream_t stream) {
  (void)in_sizes; (void)n_in; (void)out_size; (void)ws_size;
  constexpr int DIMc = 1024, DINNER = 2048, SEQ = 2048, BATCH = 2;
  constexpr int Mtok = BATCH * SEQ;  // 4096

  const float* x        = (const float*)d_in[0];
  const float* norm_w   = (const float*)d_in[1];
  const float* norm_b   = (const float*)d_in[2];
  const float* in_proj  = (const float*)d_in[3];
  const float* conv_w   = (const float*)d_in[4];
  const float* conv_b   = (const float*)d_in[5];
  const float* x_proj   = (const float*)d_in[6];
  const float* dt_proj  = (const float*)d_in[7];
  const float* dt_b     = (const float*)d_in[8];
  const float* A_log    = (const float*)d_in[9];
  const float* Dp       = (const float*)d_in[10];
  const float* out_proj = (const float*)d_in[11];
  const float* ffn_w1   = (const float*)d_in[12];
  const float* ffn_b1   = (const float*)d_in[13];
  const float* ffn_w2   = (const float*)d_in[14];
  const float* ffn_b2   = (const float*)d_in[15];
  float* out = (float*)d_out;

  char* ws = (char*)d_ws;
  size_t off = 0;
  auto alloc = [&](size_t bytes) -> char* {
    char* p = ws + off;
    off = (off + bytes + 255) & ~(size_t)255;
    return p;
  };

  __bf16* w_inproj  = (__bf16*)alloc((size_t)4096 * 1024 * 2);
  __bf16* w_xproj   = (__bf16*)alloc((size_t)96 * 2048 * 2);
  __bf16* w_dtproj  = (__bf16*)alloc((size_t)2048 * 64 * 2);
  __bf16* w_outproj = (__bf16*)alloc((size_t)1024 * 2048 * 2);
  __bf16* w_ffn1    = (__bf16*)alloc((size_t)4096 * 1024 * 2);
  __bf16* w_ffn2    = (__bf16*)alloc((size_t)1024 * 4096 * 2);
  __bf16* xln_bf    = (__bf16*)alloc((size_t)Mtok * 1024 * 2);
  float*  xz_f      = (float*) alloc((size_t)Mtok * 4096 * 4);  // reused as FFN mid f32
  float*  xc_f      = (float*) alloc((size_t)Mtok * 2048 * 4);
  __bf16* xc_bf     = (__bf16*)alloc((size_t)Mtok * 2048 * 2);
  float*  xdbc_f    = (float*) alloc((size_t)Mtok * 96 * 4);
  __bf16* dt_bf     = (__bf16*)alloc((size_t)Mtok * 64 * 2);
  float*  delta_f   = (float*) alloc((size_t)Mtok * 2048 * 4);
  __bf16* y_bf      = (__bf16*)alloc((size_t)Mtok * 2048 * 2);
  float*  h1_f      = (float*) alloc((size_t)Mtok * 1024 * 4);
  __bf16* ln2_bf    = (__bf16*)alloc((size_t)Mtok * 1024 * 2);
  __bf16* mid_bf    = (__bf16*)alloc((size_t)Mtok * 4096 * 2);

  auto cvt = [&](const float* src, __bf16* dst, int n) {
    cvt_f32_bf16_kernel<<<(n + 255) / 256, 256, 0, stream>>>(src, dst, n);
  };
  cvt(in_proj,  w_inproj,  4096 * 1024);
  cvt(x_proj,   w_xproj,   96 * 2048);
  cvt(dt_proj,  w_dtproj,  2048 * 64);
  cvt(out_proj, w_outproj, 1024 * 2048);
  cvt(ffn_w1,   w_ffn1,    4096 * 1024);
  cvt(ffn_w2,   w_ffn2,    1024 * 4096);

  // LN1
  ln_bf16_kernel<<<Mtok, 256, 0, stream>>>(x, norm_w, norm_b, xln_bf);

  // in_proj: (4096x1024) x (4096x1024)^T -> xz (4096x4096)
  gemm_bf16_kernel<<<dim3(4096 / BN, Mtok / BM), 256, 0, stream>>>(
      xln_bf, w_inproj, xz_f, Mtok, 2 * DINNER, DIMc, nullptr, nullptr);

  // causal depthwise conv + SiLU
  {
    const int n = Mtok * DINNER;
    conv_silu_kernel<<<(n + 255) / 256, 256, 0, stream>>>(
        xz_f, conv_w, conv_b, xc_f, xc_bf, SEQ, DINNER);
  }

  // x_proj: (4096x2048) x (96x2048)^T -> xdbc (4096x96)
  gemm_bf16_kernel<<<dim3(1, Mtok / BM), 256, 0, stream>>>(
      xc_bf, w_xproj, xdbc_f, Mtok, 96, DINNER, nullptr, nullptr);

  // dt_proj + bias, then softplus
  extract_dt_kernel<<<(Mtok * 64 + 255) / 256, 256, 0, stream>>>(xdbc_f, dt_bf, Mtok);
  gemm_bf16_kernel<<<dim3(2048 / BN, Mtok / BM), 256, 0, stream>>>(
      dt_bf, w_dtproj, delta_f, Mtok, DINNER, 64, dt_b, nullptr);
  softplus_kernel<<<(Mtok * 2048 + 255) / 256, 256, 0, stream>>>(delta_f, Mtok * 2048);

  // selective scan + gate
  scan_kernel<<<BATCH * (DINNER / 16), 256, 0, stream>>>(
      delta_f, xc_f, xdbc_f, xz_f, A_log, Dp, y_bf, SEQ);

  // out_proj + residual(x)
  gemm_bf16_kernel<<<dim3(1024 / BN, Mtok / BM), 256, 0, stream>>>(
      y_bf, w_outproj, h1_f, Mtok, DIMc, DINNER, nullptr, x);

  // LN2
  ln_bf16_kernel<<<Mtok, 256, 0, stream>>>(h1_f, norm_w, norm_b, ln2_bf);

  // FFN1 + bias (reuse xz_f as f32 scratch), GELU -> bf16
  gemm_bf16_kernel<<<dim3(4096 / BN, Mtok / BM), 256, 0, stream>>>(
      ln2_bf, w_ffn1, xz_f, Mtok, 4096, DIMc, ffn_b1, nullptr);
  gelu_bf16_kernel<<<(Mtok * 4096 + 255) / 256, 256, 0, stream>>>(
      xz_f, mid_bf, Mtok * 4096);

  // FFN2 + bias + residual(h1) -> d_out
  gemm_bf16_kernel<<<dim3(1024 / BN, Mtok / BM), 256, 0, stream>>>(
      mid_bf, w_ffn2, out, Mtok, DIMc, 4096, ffn_b2, h1_f);
}
